// DiffusionTransformer_1726576853224
// MI455X (gfx1250) — matrix operand order repaired and checked by reading, not compiled, not run on hardware
//
#include <hip/hip_runtime.h>
#include <math.h>

#define NTOK 384
#define DM   768
#define NH   16
#define HD_  48
#define PD   128
#define FD   1536
#define NL   24
#define NNP  (NTOK*NTOK)   // 147456

typedef __attribute__((ext_vector_type(16))) __bf16 v16bf;
typedef __attribute__((ext_vector_type(8)))  float  v8f;
typedef __attribute__((ext_vector_type(16))) unsigned short v16us;

__device__ __forceinline__ unsigned short f2bf(float f){
  unsigned int u = __float_as_uint(f);
  u += 0x7fffu + ((u >> 16) & 1u);       // round-to-nearest-even
  return (unsigned short)(u >> 16);
}
__device__ __forceinline__ float sigm(float x){ return 1.0f/(1.0f+__expf(-x)); }

// =============== generic batched bf16 WMMA GEMM =======================
// C[b, m, coloff+n] (or C[b, coloff+n, m] if CT) = sum_k A[b,m,k]*W[b,k,n] + bias[n]
// A: bf16 row-major (lda).  W: bf16, element (k,n) at  n*wn + k*wk  (wk==1 when WCONTIG).
// Wave computes 16(M) x 64(N); block = 8 waves => 64(M) x 128(N).
// Wave id is forced into an SGPR so all tile guards are scalar branches:
// EXEC stays all-1s around every WMMA (ISA requirement) and accumulators
// never get shuffled through exec-masked phi merges.
template<bool WCONTIG, bool CT>
__global__ __launch_bounds__(256) void gemm_bf16(
    const unsigned short* __restrict__ A, long aBatch, int lda,
    const unsigned short* __restrict__ W, long wBatch, long wk, long wn,
    float* __restrict__ C, long cBatch, int ldc, int coloff,
    const float* __restrict__ bias, int M, int K, int Nout)
{
  const int tid  = threadIdx.x;
  const int wid  = __builtin_amdgcn_readfirstlane(tid >> 5);   // SGPR wave id
  const int lane = tid & 31;
  const int half = lane >> 4;
  const int l16  = lane & 15;
  const int m0 = blockIdx.x*64  + (wid & 3)*16;
  const int n0 = blockIdx.y*128 + (wid >> 2)*64;
  if (m0 >= M) return;                       // scalar branch (uniform)
  const int rem = Nout - n0;
  if (rem <= 0) return;                      // whole wave strip inactive
  const int nact = rem >= 64 ? 4 : ((rem + 15) >> 4);   // SGPR

  A += (size_t)blockIdx.z*aBatch;
  W += (size_t)blockIdx.z*wBatch;
  C += (size_t)blockIdx.z*cBatch;

  v8f acc[4] = {};

  for (int k0=0; k0<K; k0+=32){
    // A fragment: lane half 0 -> K 0..15, half 1 -> K 16..31 (contiguous)
    const unsigned short* ap = A + (size_t)(m0+l16)*lda + k0 + half*16;
    v16us araw;
    *(uint4*)&araw       = *(const uint4*)(ap);
    *(((uint4*)&araw)+1) = *(const uint4*)(ap+8);
    v16bf afrag = __builtin_bit_cast(v16bf, araw);
    if (k0 + 32 < K) __builtin_prefetch((const void*)(ap + 32), 0, 1);
    #pragma unroll
    for (int nb=0; nb<4; nb++){
      if (nb >= nact) break;                 // scalar branch (uniform)
      const long n = n0 + nb*16 + l16;
      v16us braw;
      if (WCONTIG){
        // B fragment: lane half 0 -> K 0..15, half 1 -> K 16..31 (contiguous)
        const unsigned short* bp = W + n*wn + k0 + half*16;
        *(uint4*)&braw       = *(const uint4*)(bp);
        *(((uint4*)&braw)+1) = *(const uint4*)(bp+8);
        if (k0 + 32 < K) __builtin_prefetch((const void*)(bp + 32), 0, 1);
      } else {
        const unsigned short* bp = W + n*wn + (long)(k0 + half*16)*wk;
        #pragma unroll
        for (int e=0;e<16;e++) braw[e] = bp[(long)e*wk];
      }
      v16bf bfrag = __builtin_bit_cast(v16bf, braw);
      acc[nb] = __builtin_amdgcn_wmma_f32_16x16x32_bf16(
                  false, afrag, false, bfrag, (short)0, acc[nb], false, false);
    }
  }
  #pragma unroll
  for (int nb=0; nb<4; nb++){
    if (nb >= nact) break;                   // scalar branch (uniform)
    const int n = n0 + nb*16 + l16;
    const float bv = bias ? bias[n] : 0.0f;
    #pragma unroll
    for (int r=0;r<8;r++){
      const int m = m0 + half*8 + r;         // C layout: VGPR r -> M = r + 8*half
      const float v = acc[nb][r] + bv;
      if (CT) C[(size_t)(coloff+n)*ldc + m] = v;
      else    C[(size_t)m*ldc + coloff + n] = v;
    }
  }
}

// =============== elementwise / reduction helpers ======================

// per-row mean / rsqrt(var+eps); one 128-thread block per row
__global__ __launch_bounds__(128) void row_stats(
    const float* __restrict__ src, int cols, float* __restrict__ mu, float* __restrict__ rs)
{
  const int row = blockIdx.x;
  const float* p = src + (size_t)row*cols;
  float s=0.f, s2=0.f;
  for (int c=threadIdx.x; c<cols; c+=128){ float v=p[c]; s+=v; s2+=v*v; }
  __shared__ float sh0[128], sh1[128];
  sh0[threadIdx.x]=s; sh1[threadIdx.x]=s2; __syncthreads();
  for (int o=64;o>0;o>>=1){
    if (threadIdx.x<o){ sh0[threadIdx.x]+=sh0[threadIdx.x+o]; sh1[threadIdx.x]+=sh1[threadIdx.x+o]; }
    __syncthreads();
  }
  if (threadIdx.x==0){
    float m = sh0[0]/cols;
    float v = sh1[0]/cols - m*m;
    mu[row]=m; rs[row]=rsqrtf(v + 1e-5f);
  }
}

// dst_bf16 = ((src - mu[row])*rs[row]) * w[c] + b[c]   (w/b nullable)
__global__ void ln_apply(const float* __restrict__ src, const float* __restrict__ mu,
                         const float* __restrict__ rs, const float* __restrict__ w,
                         const float* __restrict__ b, unsigned short* __restrict__ dst,
                         long total, int cols)
{
  long i = (long)blockIdx.x*blockDim.x + threadIdx.x;
  if (i>=total) return;
  long row = i/cols; int c = (int)(i - row*cols);
  float v = (src[i]-mu[row])*rs[row];
  if (w) v *= w[c];
  if (b) v += b[c];
  dst[i] = f2bf(v);
}

__global__ void cvt_bf16(const float* __restrict__ src, unsigned short* __restrict__ dst, long total){
  long i = (long)blockIdx.x*blockDim.x + threadIdx.x;
  if (i<total) dst[i] = f2bf(src[i]);
}

// weight convert + transpose: src fp32 [K,N] row-major -> dst bf16 [N,K]
__global__ void convT(const float* __restrict__ src, unsigned short* __restrict__ dst, int K, int N){
  long i = (long)blockIdx.x*blockDim.x + threadIdx.x;
  if (i >= (long)K*N) return;
  int k = (int)(i / N), n = (int)(i - (long)k*N);
  dst[(size_t)n*K + k] = f2bf(src[i]);
}

// b = sigmoid(sg[:,sc_off+c]) * LN(a)[row,c] + sg[:,bi_off+c]   -> bf16
__global__ __launch_bounds__(256) void adaln_combine(
    const float* __restrict__ a, const float* __restrict__ sg, int sgld,
    int sc_off, int bi_off, unsigned short* __restrict__ outb)
{
  const int row = blockIdx.x;
  const float* ar = a + (size_t)row*DM;
  __shared__ float sh0[256], sh1[256];
  float s=0.f, s2=0.f, loc[3];
  #pragma unroll
  for (int t=0;t<3;t++){ float v = ar[threadIdx.x + t*256]; loc[t]=v; s+=v; s2+=v*v; }
  sh0[threadIdx.x]=s; sh1[threadIdx.x]=s2; __syncthreads();
  for (int o=128;o>0;o>>=1){
    if (threadIdx.x<o){ sh0[threadIdx.x]+=sh0[threadIdx.x+o]; sh1[threadIdx.x]+=sh1[threadIdx.x+o]; }
    __syncthreads();
  }
  float m = sh0[0]*(1.0f/DM);
  float var = sh1[0]*(1.0f/DM) - m*m;
  float r = rsqrtf(var + 1e-5f);
  const float* sgr = sg + (size_t)row*sgld;
  #pragma unroll
  for (int t=0;t<3;t++){
    int c = threadIdx.x + t*256;
    float an = (loc[t]-m)*r;
    outb[(size_t)row*DM + c] = f2bf(sigm(sgr[sc_off+c])*an + sgr[bi_off+c]);
  }
}

// qb/kb: [NTOK, NH*64] bf16 zero-padded (HD 48->64); vb: [NTOK, NH*48] bf16
__global__ void prep_attn(const float* __restrict__ qkvg,
                          unsigned short* __restrict__ qb, unsigned short* __restrict__ kb,
                          unsigned short* __restrict__ vb)
{
  long i = (long)blockIdx.x*blockDim.x + threadIdx.x;
  if (i >= (long)NTOK*NH*64) return;
  int row = (int)(i / (NH*64));
  int hc  = (int)(i - (long)row*(NH*64));
  int h = hc >> 6, hd = hc & 63;
  float qv=0.f, kv=0.f;
  if (hd < HD_){
    qv = qkvg[(size_t)row*3072 +        h*HD_ + hd];
    kv = qkvg[(size_t)row*3072 +  DM  + h*HD_ + hd];
    vb[(size_t)row*DM + h*HD_ + hd] = f2bf(qkvg[(size_t)row*3072 + 2*DM + h*HD_ + hd]);
  }
  qb[i] = f2bf(qv); kb[i] = f2bf(kv);
}

// softmax over j with scale, pair bias and mask; output bf16
__global__ __launch_bounds__(128) void softmax_bias(
    const float* __restrict__ S, const float* __restrict__ Bz,
    const float* __restrict__ mask, unsigned short* __restrict__ P, float scale)
{
  const int row = blockIdx.x;                 // h*NTOK + i
  const float* sp = S  + (size_t)row*NTOK;
  const float* bp = Bz + (size_t)row*NTOK;
  __shared__ float sh[128];
  float x[3]; float mx = -1e30f;
  #pragma unroll
  for (int t=0;t<3;t++){
    int j = threadIdx.x + t*128;
    float v = sp[j]*scale + bp[j] + (1.0f - mask[j]) * -1e9f;
    x[t]=v; mx=fmaxf(mx,v);
  }
  sh[threadIdx.x]=mx; __syncthreads();
  for (int o=64;o>0;o>>=1){ if (threadIdx.x<o) sh[threadIdx.x]=fmaxf(sh[threadIdx.x],sh[threadIdx.x+o]); __syncthreads(); }
  mx = sh[0]; __syncthreads();
  float sum=0.f;
  #pragma unroll
  for (int t=0;t<3;t++){ x[t]=__expf(x[t]-mx); sum+=x[t]; }
  sh[threadIdx.x]=sum; __syncthreads();
  for (int o=64;o>0;o>>=1){ if (threadIdx.x<o) sh[threadIdx.x]+=sh[threadIdx.x+o]; __syncthreads(); }
  float inv = 1.0f/sh[0];
  #pragma unroll
  for (int t=0;t<3;t++){ int j=threadIdx.x+t*128; P[(size_t)row*NTOK + j] = f2bf(x[t]*inv); }
}

// og_bf16 = sigmoid(g) * o
__global__ void gmul(const float* __restrict__ qkvg, const float* __restrict__ o,
                     unsigned short* __restrict__ og)
{
  long i = (long)blockIdx.x*blockDim.x + threadIdx.x;
  if (i >= (long)NTOK*DM) return;
  int row = (int)(i/DM), c = (int)(i - (long)row*DM);
  og[i] = f2bf(sigm(qkvg[(size_t)row*3072 + 3*DM + c]) * o[i]);
}

// a += sigmoid(sg[:, off+c]) * x
__global__ void resgate(float* __restrict__ a, const float* __restrict__ sg, int sgld,
                        int off, const float* __restrict__ x)
{
  long i = (long)blockIdx.x*blockDim.x + threadIdx.x;
  if (i >= (long)NTOK*DM) return;
  int row = (int)(i/DM), c = (int)(i - (long)row*DM);
  a[i] += sigm(sg[(size_t)row*sgld + off + c]) * x[i];
}

// h_bf16 = silu(hb[:,f]) * hb[:,FD+f]
__global__ void swiglu(const float* __restrict__ hb, unsigned short* __restrict__ h)
{
  long i = (long)blockIdx.x*blockDim.x + threadIdx.x;
  if (i >= (long)NTOK*FD) return;
  int row = (int)(i/FD), c = (int)(i - (long)row*FD);
  float u = hb[(size_t)row*(2*FD) + c];
  float w = hb[(size_t)row*(2*FD) + FD + c];
  h[i] = f2bf((u * sigm(u)) * w);
}

// combined s-side bias vector: [sc1_b | 0 | sc2_b | 0 | og_b | tg_b]
__global__ void packbias_s(const float* sc1b, const float* sc2b,
                           const float* ogb, const float* tgb, float* vec)
{
  int i = blockIdx.x*blockDim.x + threadIdx.x;
  if (i >= 6*DM) return;
  int seg = i/DM, c = i - seg*DM;
  float v = 0.f;
  if (seg==0) v=sc1b[c]; else if (seg==2) v=sc2b[c]; else if (seg==4) v=ogb[c]; else if (seg==5) v=tgb[c];
  vec[i]=v;
}
__global__ void packbias_q(const float* qb, float* vec){
  int i = blockIdx.x*blockDim.x + threadIdx.x;
  if (i >= 4*DM) return;
  vec[i] = (i<DM) ? qb[i] : 0.f;
}

// ============================ host ====================================
extern "C" void kernel_launch(void* const* d_in, const int* in_sizes, int n_in,
                              void* d_out, int out_size, void* d_ws, size_t ws_size,
                              hipStream_t stream)
{
  const float* a_in = (const float*)d_in[0];
  const float* s_in = (const float*)d_in[1];
  const float* z_in = (const float*)d_in[2];
  const float* mask = (const float*)d_in[3];
  const float* ada1_sn_w    = (const float*)d_in[4];
  const float* ada1_scale_w = (const float*)d_in[5];
  const float* ada1_scale_b = (const float*)d_in[6];
  const float* ada1_bias_w  = (const float*)d_in[7];
  const float* zn_w = (const float*)d_in[8];
  const float* zn_b = (const float*)d_in[9];
  const float* z_w  = (const float*)d_in[10];
  const float* q_w  = (const float*)d_in[11];
  const float* q_b  = (const float*)d_in[12];
  const float* k_w  = (const float*)d_in[13];
  const float* v_w  = (const float*)d_in[14];
  const float* g_w  = (const float*)d_in[15];
  const float* o_w  = (const float*)d_in[16];
  const float* og_w = (const float*)d_in[17];
  const float* og_b = (const float*)d_in[18];
  const float* ada2_sn_w    = (const float*)d_in[19];
  const float* ada2_scale_w = (const float*)d_in[20];
  const float* ada2_scale_b = (const float*)d_in[21];
  const float* ada2_bias_w  = (const float*)d_in[22];
  const float* sw_w = (const float*)d_in[23];
  const float* ab_w = (const float*)d_in[24];
  const float* ba_w = (const float*)d_in[25];
  const float* tg_w = (const float*)d_in[26];
  const float* tg_b = (const float*)d_in[27];

  // ---- workspace bump allocator ----
  char* wsb = (char*)d_ws; size_t off = 0;
  auto alloc = [&](size_t bytes)->void*{ void* p = wsb+off; off = (off+bytes+255) & ~(size_t)255; return p; };
  float*  mu_z = (float*)alloc(sizeof(float)*NNP);
  float*  rs_z = (float*)alloc(sizeof(float)*NNP);
  float*  mu_s = (float*)alloc(sizeof(float)*NTOK);
  float*  rs_s = (float*)alloc(sizeof(float)*NTOK);
  unsigned short* zb   = (unsigned short*)alloc(2ull*NNP*PD);
  float*  biasS = (float*)alloc(sizeof(float)*(size_t)NH*NNP);
  float*  Smat  = (float*)alloc(sizeof(float)*(size_t)NH*NNP);
  unsigned short* Pmat = (unsigned short*)alloc(2ull*NH*NNP);
  unsigned short* sraw = (unsigned short*)alloc(2ull*NTOK*DM);
  unsigned short* sln  = (unsigned short*)alloc(2ull*NTOK*DM);
  float*  sgate = (float*)alloc(sizeof(float)*(size_t)NTOK*6*DM);
  float*  bvs   = (float*)alloc(sizeof(float)*6*DM);
  float*  bvq   = (float*)alloc(sizeof(float)*4*DM);
  unsigned short* bbuf = (unsigned short*)alloc(2ull*NTOK*DM);
  float*  qkvg  = (float*)alloc(sizeof(float)*(size_t)NTOK*4*DM);
  unsigned short* qb = (unsigned short*)alloc(2ull*NTOK*NH*64);
  unsigned short* kb = (unsigned short*)alloc(2ull*NTOK*NH*64);
  unsigned short* vb = (unsigned short*)alloc(2ull*NTOK*DM);
  float*  obuf  = (float*)alloc(sizeof(float)*(size_t)NTOK*DM);
  unsigned short* ogb16 = (unsigned short*)alloc(2ull*NTOK*DM);
  float*  xbuf  = (float*)alloc(sizeof(float)*(size_t)NTOK*DM);       // gemm-o / gemm-ba output
  float*  hbraw = (float*)alloc(sizeof(float)*(size_t)NTOK*2*FD);
  unsigned short* hbf = (unsigned short*)alloc(2ull*NTOK*FD);
  unsigned short* WtS    = (unsigned short*)alloc(2ull*6*DM*DM);
  unsigned short* WtQKVG = (unsigned short*)alloc(2ull*4*DM*DM);
  unsigned short* WtO    = (unsigned short*)alloc(2ull*DM*DM);
  unsigned short* WtSWAB = (unsigned short*)alloc(2ull*2*FD*DM);
  unsigned short* WtBA   = (unsigned short*)alloc(2ull*DM*FD);
  unsigned short* zwT    = (unsigned short*)alloc(2ull*NH*PD);
  if (off > ws_size) return;  // workspace too small: bail safely

  const float scale = 1.0f/sqrtf((float)HD_);
  float* a = (float*)d_out;   // residual stream lives in the output buffer

  // a <- a_in; layer-invariant precompute
  hipMemcpyAsync(a, a_in, sizeof(float)*(size_t)NTOK*DM, hipMemcpyDeviceToDevice, stream);
  row_stats<<<NNP, 128, 0, stream>>>(z_in, PD, mu_z, rs_z);
  row_stats<<<NTOK,128, 0, stream>>>(s_in, DM, mu_s, rs_s);
  cvt_bf16<<<(NTOK*DM+255)/256, 256, 0, stream>>>(s_in, sraw, (long)NTOK*DM);

  auto cvtW = [&](const float* src, unsigned short* dst, int K, int N){
    long tot = (long)K*N;
    convT<<<(unsigned)((tot+255)/256), 256, 0, stream>>>(src, dst, K, N);
  };

  for (int l=0; l<NL; l++){
    const size_t dd = (size_t)l*DM*DM, df = (size_t)l*DM*FD, d1 = (size_t)l*DM;
    // --- weight conversion to bf16 [N,K] (transposed) ---
    cvtW(ada1_scale_w+dd, WtS + 0ull*DM*DM, DM, DM);
    cvtW(ada1_bias_w +dd, WtS + 1ull*DM*DM, DM, DM);
    cvtW(ada2_scale_w+dd, WtS + 2ull*DM*DM, DM, DM);
    cvtW(ada2_bias_w +dd, WtS + 3ull*DM*DM, DM, DM);
    cvtW(og_w+dd,         WtS + 4ull*DM*DM, DM, DM);
    cvtW(tg_w+dd,         WtS + 5ull*DM*DM, DM, DM);
    cvtW(q_w+dd, WtQKVG + 0ull*DM*DM, DM, DM);
    cvtW(k_w+dd, WtQKVG + 1ull*DM*DM, DM, DM);
    cvtW(v_w+dd, WtQKVG + 2ull*DM*DM, DM, DM);
    cvtW(g_w+dd, WtQKVG + 3ull*DM*DM, DM, DM);
    cvtW(o_w+dd, WtO, DM, DM);
    cvtW(sw_w+df, WtSWAB + 0ull,            DM, FD);
    cvtW(ab_w+df, WtSWAB + (size_t)FD*DM,   DM, FD);
    cvtW(ba_w+df, WtBA, FD, DM);
    cvtW(z_w + (size_t)l*PD*NH, zwT, PD, NH);
    packbias_s<<<(6*DM+255)/256,256,0,stream>>>(ada1_scale_b+d1, ada2_scale_b+d1, og_b+d1, tg_b+d1, bvs);
    packbias_q<<<(4*DM+255)/256,256,0,stream>>>(q_b+d1, bvq);

    // --- s-conditioned projections: [sc1,bi1 | sc2,bi2 | og,tg] into sgate[384,4608] ---
    ln_apply<<<(NTOK*DM+255)/256,256,0,stream>>>(s_in, mu_s, rs_s, ada1_sn_w+d1, nullptr, sln, (long)NTOK*DM, DM);
    gemm_bf16<true,false><<<dim3(NTOK/64,(2*DM+127)/128,1),256,0,stream>>>(
        sln,0,DM, WtS,0,1,DM, sgate,0,6*DM,0, bvs, NTOK,DM,2*DM);
    ln_apply<<<(NTOK*DM+255)/256,256,0,stream>>>(s_in, mu_s, rs_s, ada2_sn_w+d1, nullptr, sln, (long)NTOK*DM, DM);
    gemm_bf16<true,false><<<dim3(NTOK/64,(2*DM+127)/128,1),256,0,stream>>>(
        sln,0,DM, WtS+(size_t)2*DM*DM,0,1,DM, sgate,0,6*DM,2*DM, bvs+2*DM, NTOK,DM,2*DM);
    gemm_bf16<true,false><<<dim3(NTOK/64,(2*DM+127)/128,1),256,0,stream>>>(
        sraw,0,DM, WtS+(size_t)4*DM*DM,0,1,DM, sgate,0,6*DM,4*DM, bvs+4*DM, NTOK,DM,2*DM);

    // --- AdaLN #1 + fused q,k,v,g projection ---
    adaln_combine<<<NTOK,256,0,stream>>>(a, sgate, 6*DM, 0, DM, bbuf);
    gemm_bf16<true,false><<<dim3(NTOK/64,(4*DM+127)/128,1),256,0,stream>>>(
        bbuf,0,DM, WtQKVG,0,1,DM, qkvg,0,4*DM,0, bvq, NTOK,DM,4*DM);
    prep_attn<<<(NTOK*NH*64+255)/256,256,0,stream>>>(qkvg, qb, kb, vb);

    // --- pair bias: LN(z)[NN,128] @ z_wT -> biasS[H,N,N] (transposed store) ---
    ln_apply<<<(unsigned)(((long)NNP*PD+255)/256),256,0,stream>>>(
        z_in, mu_z, rs_z, zn_w+(size_t)l*PD, zn_b+(size_t)l*PD, zb, (long)NNP*PD, PD);
    gemm_bf16<true,true><<<dim3(NNP/64,1,1),256,0,stream>>>(
        zb,0,PD, zwT,0,1,PD, biasS,0,NNP,0, nullptr, NNP,PD,NH);

    // --- attention: S = Q K^T (batched over heads, K=64 zero-padded) ---
    gemm_bf16<true,false><<<dim3(NTOK/64,(NTOK+127)/128,NH),256,0,stream>>>(
        qb,64,NH*64, kb,64,1,NH*64, Smat,NNP,NTOK,0, nullptr, NTOK,64,NTOK);
    softmax_bias<<<NH*NTOK,128,0,stream>>>(Smat, biasS, mask, Pmat, scale);
    // O = P V (W strided: element (k=j, n=hd) at j*DM + h*48 + n)
    gemm_bf16<false,false><<<dim3(NTOK/64,1,NH),256,0,stream>>>(
        Pmat,NNP,NTOK, vb,HD_,DM,1, obuf,HD_,DM,0, nullptr, NTOK,NTOK,HD_);

    // --- gate, output projection, residual ---
    gmul<<<(NTOK*DM+255)/256,256,0,stream>>>(qkvg, obuf, ogb16);
    gemm_bf16<true,false><<<dim3(NTOK/64,(DM+127)/128,1),256,0,stream>>>(
        ogb16,0,DM, WtO,0,1,DM, xbuf,0,DM,0, nullptr, NTOK,DM,DM);
    resgate<<<(NTOK*DM+255)/256,256,0,stream>>>(a, sgate, 6*DM, 4*DM, xbuf);

    // --- conditioned transition (SwiGLU) ---
    adaln_combine<<<NTOK,256,0,stream>>>(a, sgate, 6*DM, 2*DM, 3*DM, bbuf);
    gemm_bf16<true,false><<<dim3(NTOK/64,(2*FD+127)/128,1),256,0,stream>>>(
        bbuf,0,DM, WtSWAB,0,1,DM, hbraw,0,2*FD,0, nullptr, NTOK,DM,2*FD);
    swiglu<<<(NTOK*FD+255)/256,256,0,stream>>>(hbraw, hbf);
    gemm_bf16<true,false><<<dim3(NTOK/64,(DM+127)/128,1),256,0,stream>>>(
        hbf,0,FD, WtBA,0,1,FD, xbuf,0,DM,0, nullptr, NTOK,FD,DM);
    resgate<<<(NTOK*DM+255)/256,256,0,stream>>>(a, sgate, 6*DM, 5*DM, xbuf);
  }
}
